// TransformerEncoderMoE_62560493633926
// MI455X (gfx1250) — matrix operand order repaired and checked by reading, not compiled, not run on hardware
//
#include <hip/hip_runtime.h>
#include <math.h>

// Problem constants (from the reference)
#define L_  2
#define D_  1024
#define H_  16
#define FF_ 4096
#define E_  8
#define S_  1024
#define B_  8
#define HD_ 64          // D/H
#define T_  8192        // S*B

typedef __attribute__((ext_vector_type(16))) _Float16 v16h;
typedef __attribute__((ext_vector_type(8)))  float    v8f;

static __device__ inline v8f zero8() {
  v8f z = {0.f,0.f,0.f,0.f,0.f,0.f,0.f,0.f};
  return z;
}

static __device__ inline v8f wmma16(v16h a, v16h b, v8f c) {
  // D(f32 16x16) = A(f16 16x32) * B(f16 32x16) + C
  return __builtin_amdgcn_wmma_f32_16x16x32_f16(
      false, a, false, b, (short)0, c, false, false);
}

// ---- Fragment loaders (layouts per CDNA5 ISA 7.12.2) -----------------------
// A fragment 16x32 (M x K), element e of lane L:
//   m = L&15, half = L>>4,  K = 16*(e>>3) + 8*half + (e&7)
static __device__ inline v16h load_a_frag_g(const float* A, size_t lda, int lane) {
  int m = lane & 15, hf = lane >> 4;
  const float* r = A + (size_t)m * lda;
  v16h out;
#pragma unroll
  for (int e = 0; e < 16; ++e) {
    int k = ((e >> 3) << 4) + (hf << 3) + (e & 7);
    out[e] = (_Float16)r[k];
  }
  return out;
}

static __device__ inline v16h load_a_frag_lds(const _Float16* A, int ldh, int lane) {
  int m = lane & 15, hf = lane >> 4;
  const _Float16* r = A + m * ldh;
  v16h out;
#pragma unroll
  for (int e = 0; e < 16; ++e) {
    int k = ((e >> 3) << 4) + (hf << 3) + (e & 7);
    out[e] = r[k];
  }
  return out;
}

// A fragment from f32 tile resident in LDS (filled by async copies)
static __device__ inline v16h load_a_frag_lds_f32(const float* A, int ldf, int lane) {
  int m = lane & 15, hf = lane >> 4;
  const float* r = A + m * ldf;
  v16h out;
#pragma unroll
  for (int e = 0; e < 16; ++e) {
    int k = ((e >> 3) << 4) + (hf << 3) + (e & 7);
    out[e] = (_Float16)r[k];
  }
  return out;
}

// B fragment 32x16 (K x N), W stored row-major [N][K'] (row = output col n):
//   n = L&15, half = L>>4, K = 16*(e>>3) + 8*half + (e&7)
static __device__ inline v16h load_b_frag_g(const float* W, size_t ldb, int lane) {
  int n = lane & 15, hf = lane >> 4;
  const float* r = W + (size_t)n * ldb + (hf << 3);
  v16h out;
#pragma unroll
  for (int e = 0; e < 16; ++e) out[e] = (_Float16)r[e + ((e >> 3) << 3)];
  return out;
}

// B fragment where memory is K-major (rows = K with stride rs, col = n)
static __device__ inline v16h load_bT_frag_g(const float* P, size_t rs, int lane) {
  int n = lane & 15, hf = lane >> 4;
  const float* p = P + (size_t)(hf * 8) * rs + n;
  v16h out;
#pragma unroll
  for (int e = 0; e < 16; ++e) out[e] = (_Float16)p[(size_t)(e + ((e >> 3) << 3)) * rs];
  return out;
}

// ---- Elementwise add: qk_in = x + pos --------------------------------------
__global__ void k_add(const float* __restrict__ a, const float* __restrict__ b,
                      float* __restrict__ c, int n) {
  int i = blockIdx.x * blockDim.x + threadIdx.x;
  if (i < n) c[i] = a[i] + b[i];
}

// ---- Generic GEMM: C[M,N] = A[M,K] * W[N,K]^T + bias[N] --------------------
// block 256 (8 waves), WG tile 64(M) x 128(N); wave w owns 16 N-cols.
// A tile double-buffered in LDS via GLOBAL_LOAD_ASYNC_TO_LDS_B128 (ASYNCcnt),
// overlapping the k-step i+1 copy with the k-step i WMMAs.
__global__ __launch_bounds__(256) void k_gemm(
    const float* __restrict__ A, const float* __restrict__ W,
    const float* __restrict__ bias, float* __restrict__ C,
    int M, int N, int Kd) {
  __shared__ float as[2][64 * 36];
  const int ALDF = 36;                 // f32 pitch; gcd(36,64)=4 -> 16 banks/frag
  int tid = threadIdx.x, lane = tid & 31, w = tid >> 5, hf = lane >> 4;
  int m0 = blockIdx.x * 64, n0 = blockIdx.y * 128;

  v8f cacc[4];
#pragma unroll
  for (int tb = 0; tb < 4; ++tb) cacc[tb] = zero8();

  // two 16B chunks per thread per 64x32 tile (512 chunks total)
  int cA = tid,        rowA = cA >> 3, kcA = (cA & 7) * 4;
  int cB = tid + 256,  rowB = cB >> 3, kcB = (cB & 7) * 4;
  unsigned ldsA0 = (unsigned)(size_t)&as[0][rowA * ALDF + kcA];
  unsigned ldsB0 = (unsigned)(size_t)&as[0][rowB * ALDF + kcB];
  unsigned ldsA1 = (unsigned)(size_t)&as[1][rowA * ALDF + kcA];
  unsigned ldsB1 = (unsigned)(size_t)&as[1][rowB * ALDF + kcB];
  const float* gA = A + (size_t)(m0 + rowA) * Kd + kcA;
  const float* gB = A + (size_t)(m0 + rowB) * Kd + kcB;

#define ISSUE_ASYNC(kkv, buf)                                                  \
  do {                                                                         \
    unsigned long long a0 = (unsigned long long)(size_t)(gA + (kkv));          \
    unsigned long long a1 = (unsigned long long)(size_t)(gB + (kkv));          \
    unsigned d0 = (buf) ? ldsA1 : ldsA0;                                       \
    unsigned d1 = (buf) ? ldsB1 : ldsB0;                                       \
    asm volatile("global_load_async_to_lds_b128 %0, %1, off"                   \
                 :: "v"(d0), "v"(a0) : "memory");                              \
    asm volatile("global_load_async_to_lds_b128 %0, %1, off"                   \
                 :: "v"(d1), "v"(a1) : "memory");                              \
  } while (0)

  ISSUE_ASYNC(0, 0);

  const float* wbase = W + (size_t)(n0 + w * 16) * Kd;
  int nsteps = Kd >> 5;
  for (int i = 0; i < nsteps; ++i) {
    int kk = i << 5;
    if (i + 1 < nsteps) {
      ISSUE_ASYNC(kk + 32, (i + 1) & 1);
      // wait for this step's 2 copies (in-order completion), keep next 2 in flight
      asm volatile("s_wait_asynccnt 0x2" ::: "memory");
    } else {
      asm volatile("s_wait_asynccnt 0x0" ::: "memory");
    }
    __syncthreads();   // all waves' copies for this step visible

    v16h bf = load_b_frag_g(wbase + kk, (size_t)Kd, lane);
    const float* abuf = as[i & 1];
#pragma unroll
    for (int tb = 0; tb < 4; ++tb) {
      v16h a = load_a_frag_lds_f32(abuf + tb * 16 * ALDF, ALDF, lane);
      cacc[tb] = wmma16(a, bf, cacc[tb]);
    }
    __syncthreads();   // readers done before this buffer is overwritten
  }
#undef ISSUE_ASYNC

  int n = n0 + w * 16 + (lane & 15);
  float bv = bias[n];
#pragma unroll
  for (int tb = 0; tb < 4; ++tb) {
#pragma unroll
    for (int r = 0; r < 8; ++r) {
      int m = m0 + tb * 16 + hf * 8 + r;
      C[(size_t)m * N + n] = cacc[tb][r] + bv;
    }
  }
}

// ---- Attention: softmax(QK^T/8)V per (b,h); 1 wave handles 16 queries ------
__global__ __launch_bounds__(32) void k_attn(
    const float* __restrict__ q, const float* __restrict__ k,
    const float* __restrict__ v, float* __restrict__ o) {
  __shared__ float    sc[16 * 1028];
  __shared__ _Float16 pb[16 * 1032];
  const int SLD = 1028, PLD = 1032;
  int lane = threadIdx.x;
  int hf = lane >> 4;
  int s0 = blockIdx.x * 16;
  int bh = blockIdx.y;
  int b = bh % B_, h = bh / B_;
  size_t rs = (size_t)B_ * D_;      // row stride between sequence positions

  const float* qbase = q + ((size_t)s0 * B_ + b) * D_ + h * HD_;
  const float* kbase = k + (size_t)b * D_ + h * HD_;
  const float* vbase = v + (size_t)b * D_ + h * HD_;

  // Pass 1: scores = Q*K^T / 8  into LDS
  for (int kt = 0; kt < S_ / 16; ++kt) {
    v8f acc = zero8();
#pragma unroll
    for (int d0 = 0; d0 < HD_; d0 += 32) {
      v16h a  = load_a_frag_g(qbase + d0, rs, lane);
      v16h bf = load_b_frag_g(kbase + (size_t)(kt * 16) * rs + d0, rs, lane);
      acc = wmma16(a, bf, acc);
    }
    int col = kt * 16 + (lane & 15);
#pragma unroll
    for (int r = 0; r < 8; ++r) sc[(hf * 8 + r) * SLD + col] = acc[r] * 0.125f;
  }
  __syncthreads();

  // Pass 2: row softmax (lane pair {L, L^16} shares a row, halves of 512 cols)
  {
    int row = lane & 15;
    int c0 = hf * 512;
    float mx = -3.0e38f;
    for (int c = 0; c < 512; ++c) mx = fmaxf(mx, sc[row * SLD + c0 + c]);
    mx = fmaxf(mx, __shfl_xor(mx, 16));
    float sm = 0.f;
    for (int c = 0; c < 512; ++c) {
      float e = __expf(sc[row * SLD + c0 + c] - mx);
      sc[row * SLD + c0 + c] = e;
      sm += e;
    }
    sm += __shfl_xor(sm, 16);
    float inv = 1.f / sm;
    for (int c = 0; c < 512; ++c)
      pb[row * PLD + c0 + c] = (_Float16)(sc[row * SLD + c0 + c] * inv);
  }
  __syncthreads();

  // Pass 3: O = P * V
  v8f oc[4];
#pragma unroll
  for (int j = 0; j < 4; ++j) oc[j] = zero8();
  for (int kt = 0; kt < S_; kt += 32) {
    v16h a = load_a_frag_lds(pb + kt, PLD, lane);
#pragma unroll
    for (int j = 0; j < 4; ++j) {
      v16h bf = load_bT_frag_g(vbase + (size_t)kt * rs + j * 16, rs, lane);
      oc[j] = wmma16(a, bf, oc[j]);
    }
  }
  float* obase = o + ((size_t)s0 * B_ + b) * D_ + h * HD_;
#pragma unroll
  for (int j = 0; j < 4; ++j) {
    int d = j * 16 + (lane & 15);
#pragma unroll
    for (int r = 0; r < 8; ++r) obase[(size_t)(hf * 8 + r) * rs + d] = oc[j][r];
  }
}

// ---- LayerNorm: out = LN(xin + delta) * w + b, one token per block ---------
__global__ __launch_bounds__(256) void k_ln(
    const float* __restrict__ xin, const float* __restrict__ delta,
    const float* __restrict__ w, const float* __restrict__ b,
    float* __restrict__ out) {
  __shared__ float xs[D_];
  __shared__ float red[9];
  int t = blockIdx.x, tid = threadIdx.x;
  size_t base = (size_t)t * D_;

  float s = 0.f;
  for (int i = tid; i < D_; i += 256) {
    float v = xin[base + i] + delta[base + i];
    xs[i] = v;
    s += v;
  }
#pragma unroll
  for (int off = 16; off > 0; off >>= 1) s += __shfl_xor(s, off);
  if ((tid & 31) == 0) red[tid >> 5] = s;
  __syncthreads();
  if (tid == 0) {
    float tot = 0.f;
    for (int i = 0; i < 8; ++i) tot += red[i];
    red[8] = tot;
  }
  __syncthreads();
  float mean = red[8] * (1.f / D_);

  float s2 = 0.f;
  for (int i = tid; i < D_; i += 256) {
    float d = xs[i] - mean;
    s2 += d * d;
  }
#pragma unroll
  for (int off = 16; off > 0; off >>= 1) s2 += __shfl_xor(s2, off);
  __syncthreads();
  if ((tid & 31) == 0) red[tid >> 5] = s2;
  __syncthreads();
  if (tid == 0) {
    float tot = 0.f;
    for (int i = 0; i < 8; ++i) tot += red[i];
    red[8] = tot;
  }
  __syncthreads();
  float inv = rsqrtf(red[8] * (1.f / D_) + 1e-5f);

  for (int i = tid; i < D_; i += 256)
    out[base + i] = (xs[i] - mean) * inv * w[i] + b[i];
}

// ---- Gate: top-2 softmax routing weights, usage accumulation ---------------
__global__ __launch_bounds__(32) void k_gate(
    const float* __restrict__ x, const float* __restrict__ gw,
    const float* __restrict__ gb, float* __restrict__ cw,
    float* __restrict__ usage) {
  int t = blockIdx.x, lane = threadIdx.x;
  float acc[E_];
#pragma unroll
  for (int e = 0; e < E_; ++e) acc[e] = 0.f;
  for (int d = lane; d < D_; d += 32) {
    float xv = x[(size_t)t * D_ + d];
#pragma unroll
    for (int e = 0; e < E_; ++e) acc[e] += xv * gw[e * D_ + d];
  }
#pragma unroll
  for (int e = 0; e < E_; ++e) {
#pragma unroll
    for (int off = 16; off > 0; off >>= 1) acc[e] += __shfl_xor(acc[e], off);
  }
  if (lane == 0) {
    float s[E_];
#pragma unroll
    for (int e = 0; e < E_; ++e) s[e] = acc[e] + gb[e];
    int i1 = 0;
    for (int e = 1; e < E_; ++e) if (s[e] > s[i1]) i1 = e;
    int i2 = -1;
    for (int e = 0; e < E_; ++e)
      if (e != i1 && (i2 < 0 || s[e] > s[i2])) i2 = e;
    float e2 = __expf(s[i2] - s[i1]);
    float den = 1.f + e2;
    float p1 = 1.f / den, p2 = e2 / den;
#pragma unroll
    for (int e = 0; e < E_; ++e)
      cw[(size_t)t * E_ + e] = (e == i1) ? p1 : ((e == i2) ? p2 : 0.f);
    atomicAdd(&usage[i1], p1);
    atomicAdd(&usage[i2], p2);
  }
}

// ---- Aux loss accumulation (entropy of expert usage) -----------------------
__global__ void k_aux(const float* __restrict__ usage, float* __restrict__ aux_out,
                      int first) {
  float s = 0.f, u[E_];
#pragma unroll
  for (int e = 0; e < E_; ++e) { u[e] = usage[e]; s += u[e]; }
  float a = 0.f;
#pragma unroll
  for (int e = 0; e < E_; ++e) {
    float p = u[e] / s;
    a -= p * logf(p + 1e-9f);
  }
  a *= (1.f / (float)L_);
  if (first) *aux_out = a; else *aux_out += a;
}

// ---- Fused MoE expert: out += cw[:,e] * (relu(x*w1^T+b1)*w2^T + b2) --------
// block 256 (8 waves), 32 tokens x expert per WG. Hidden chunk (32x128) in LDS.
__global__ __launch_bounds__(256) void k_moe(
    const float* __restrict__ x, const float* __restrict__ w1,
    const float* __restrict__ b1, const float* __restrict__ w2,
    const float* __restrict__ b2, const float* __restrict__ cw,
    float* __restrict__ out) {
  __shared__ _Float16 xs[32 * 1032];
  __shared__ _Float16 hs[32 * 136];
  __shared__ float    cws[32];
  const int XLD = 1032, HLD = 136;
  int tid = threadIdx.x, lane = tid & 31, w = tid >> 5, hf = lane >> 4;
  int t0 = blockIdx.x * 32;
  int e  = blockIdx.y;

  const float* w1e = w1 + (size_t)e * FF_ * D_;
  const float* b1e = b1 + (size_t)e * FF_;
  const float* w2e = w2 + (size_t)e * D_ * FF_;
  const float* b2e = b2 + (size_t)e * D_;

  {
    int row = tid >> 3, seg = (tid & 7) * 128;
    const float* src = x + (size_t)(t0 + row) * D_ + seg;
#pragma unroll 8
    for (int i = 0; i < 128; ++i) xs[row * XLD + seg + i] = (_Float16)src[i];
  }
  if (tid < 32) cws[tid] = cw[(size_t)(t0 + tid) * E_ + e];
  __syncthreads();

  v8f cacc[2][8];
#pragma unroll
  for (int tb = 0; tb < 2; ++tb)
#pragma unroll
    for (int nb = 0; nb < 8; ++nb) cacc[tb][nb] = zero8();

  for (int f0 = 0; f0 < FF_; f0 += 128) {
    // Phase A: wave w computes h block [32 x 16] for f-cols f0 + w*16 ..
    v8f h0 = zero8(), h1 = zero8();
    const float* w1row = w1e + (size_t)(f0 + w * 16) * D_;
    for (int kk = 0; kk < D_; kk += 32) {
      v16h bf = load_b_frag_g(w1row + kk, (size_t)D_, lane);
      v16h a0 = load_a_frag_lds(xs + kk, XLD, lane);
      v16h a1 = load_a_frag_lds(xs + 16 * XLD + kk, XLD, lane);
      h0 = wmma16(a0, bf, h0);
      h1 = wmma16(a1, bf, h1);
    }
    float bb = b1e[f0 + w * 16 + (lane & 15)];
    int colh = w * 16 + (lane & 15);
#pragma unroll
    for (int r = 0; r < 8; ++r) {
      float v0 = h0[r] + bb; v0 = v0 > 0.f ? v0 : 0.f;
      float v1 = h1[r] + bb; v1 = v1 > 0.f ? v1 : 0.f;
      hs[(hf * 8 + r) * HLD + colh]        = (_Float16)v0;
      hs[(16 + hf * 8 + r) * HLD + colh]   = (_Float16)v1;
    }
    __syncthreads();

    // Phase B: wave w accumulates y for its 128 output cols over this f chunk
#pragma unroll
    for (int s = 0; s < 4; ++s) {
      v16h a0 = load_a_frag_lds(hs + s * 32, HLD, lane);
      v16h a1 = load_a_frag_lds(hs + 16 * HLD + s * 32, HLD, lane);
#pragma unroll
      for (int nb = 0; nb < 8; ++nb) {
        v16h bf = load_b_frag_g(
            w2e + (size_t)(w * 128 + nb * 16) * FF_ + f0 + s * 32,
            (size_t)FF_, lane);
        cacc[0][nb] = wmma16(a0, bf, cacc[0][nb]);
        cacc[1][nb] = wmma16(a1, bf, cacc[1][nb]);
      }
    }
    __syncthreads();
  }

#pragma unroll
  for (int nb = 0; nb < 8; ++nb) {
    int d = w * 128 + nb * 16 + (lane & 15);
    float b2v = b2e[d];
#pragma unroll
    for (int tb = 0; tb < 2; ++tb) {
#pragma unroll
      for (int r = 0; r < 8; ++r) {
        int m = tb * 16 + hf * 8 + r;
        float val = (cacc[tb][nb][r] + b2v) * cws[m];
        atomicAdd(&out[(size_t)(t0 + m) * D_ + d], val);
      }
    }
  }
}

// ---------------------------------------------------------------------------
extern "C" void kernel_launch(void* const* d_in, const int* in_sizes, int n_in,
                              void* d_out, int out_size, void* d_ws, size_t ws_size,
                              hipStream_t stream) {
  (void)in_sizes; (void)n_in; (void)out_size; (void)ws_size;
  const float* src    = (const float*)d_in[0];
  const float* pos    = (const float*)d_in[1];
  const float* qkv_w  = (const float*)d_in[2];
  const float* qkv_b  = (const float*)d_in[3];
  const float* out_w  = (const float*)d_in[4];
  const float* out_b  = (const float*)d_in[5];
  const float* ln1_w  = (const float*)d_in[6];
  const float* ln1_b  = (const float*)d_in[7];
  const float* ln2_w  = (const float*)d_in[8];
  const float* ln2_b  = (const float*)d_in[9];
  const float* gate_w = (const float*)d_in[10];
  const float* gate_b = (const float*)d_in[11];
  const float* w1     = (const float*)d_in[12];
  const float* b1     = (const float*)d_in[13];
  const float* w2     = (const float*)d_in[14];
  const float* b2     = (const float*)d_in[15];
  float* out = (float*)d_out;

  const size_t TD = (size_t)T_ * D_;
  float* wsf   = (float*)d_ws;
  float* W0    = wsf + 0 * TD;   // qk_in / x_mid
  float* Wq    = wsf + 1 * TD;   // q, then moe_out
  float* Wk    = wsf + 2 * TD;   // k
  float* Wv    = wsf + 3 * TD;   // v
  float* Wo    = wsf + 4 * TD;   // attention output
  float* Wp    = wsf + 5 * TD;   // attention projection
  float* Wx    = wsf + 6 * TD;   // x after layer 0
  float* cw    = wsf + 7 * TD;   // [T, E]
  float* usage = cw + (size_t)T_ * E_;

  dim3 gemm_grid(T_ / 64, D_ / 128);

  for (int l = 0; l < L_; ++l) {
    const float* x_in = (l == 0) ? src : Wx;
    const float* qkvW = qkv_w + (size_t)l * 3 * D_ * D_;
    const float* qkvB = qkv_b + (size_t)l * 3 * D_;

    // qk_in = x + pos
    k_add<<<(int)(TD / 256), 256, 0, stream>>>(x_in, pos, W0, (int)TD);

    // Q, K from qk_in ; V from x
    k_gemm<<<gemm_grid, 256, 0, stream>>>(W0, qkvW,                 qkvB,          Wq, T_, D_, D_);
    k_gemm<<<gemm_grid, 256, 0, stream>>>(W0, qkvW + (size_t)D_*D_, qkvB + D_,     Wk, T_, D_, D_);
    k_gemm<<<gemm_grid, 256, 0, stream>>>(x_in, qkvW + (size_t)2*D_*D_, qkvB + 2*D_, Wv, T_, D_, D_);

    // attention
    k_attn<<<dim3(S_ / 16, B_ * H_), 32, 0, stream>>>(Wq, Wk, Wv, Wo);

    // out projection
    k_gemm<<<gemm_grid, 256, 0, stream>>>(Wo, out_w + (size_t)l * D_ * D_,
                                          out_b + (size_t)l * D_, Wp, T_, D_, D_);

    // x_mid = LN(x + attn)
    k_ln<<<T_, 256, 0, stream>>>(x_in, Wp, ln1_w + (size_t)l * D_,
                                 ln1_b + (size_t)l * D_, W0);

    // MoE
    hipMemsetAsync(usage, 0, E_ * sizeof(float), stream);
    hipMemsetAsync(Wq, 0, TD * sizeof(float), stream);   // moe accumulator
    k_gate<<<T_, 32, 0, stream>>>(W0, gate_w + (size_t)l * E_ * D_,
                                  gate_b + (size_t)l * E_, cw, usage);
    k_aux<<<1, 1, 0, stream>>>(usage, out + TD, (l == 0) ? 1 : 0);
    k_moe<<<dim3(T_ / 32, E_), 256, 0, stream>>>(
        W0, w1 + (size_t)l * E_ * FF_ * D_, b1 + (size_t)l * E_ * FF_,
        w2 + (size_t)l * E_ * D_ * FF_, b2 + (size_t)l * E_ * D_, cw, Wq);

    // x = LN(x_mid + moe)
    float* xo = (l == 0) ? Wx : out;
    k_ln<<<T_, 256, 0, stream>>>(W0, Wq, ln2_w + (size_t)l * D_,
                                 ln2_b + (size_t)l * D_, xo);
  }
}
